// Network_19524921328061
// MI455X (gfx1250) — matrix operand order repaired and checked by reading, not compiled, not run on hardware
//
#include <hip/hip_runtime.h>

typedef __attribute__((ext_vector_type(16))) __bf16 v16bf;
typedef __attribute__((ext_vector_type(8)))  float  v8f;
typedef int v4i __attribute__((vector_size(4 * sizeof(int))));
typedef __attribute__((address_space(1))) v4i g_v4i;   // global
typedef __attribute__((address_space(3))) v4i l_v4i;   // LDS

#if defined(__AMDGCN__) && __has_builtin(__builtin_amdgcn_global_load_async_to_lds_b128)
#define USE_ASYNC_LDS 1
#else
#define USE_ASYNC_LDS 0
#endif

__device__ __forceinline__ unsigned short f2bf(float f) {
    unsigned int u = __float_as_uint(f);
    u += 0x7FFFu + ((u >> 16) & 1u);          // round-to-nearest-even
    return (unsigned short)(u >> 16);
}

// X (fp32) -> bf16
__global__ void __launch_bounds__(256) k_cvt_x(const float* __restrict__ x,
                                               unsigned short* __restrict__ out,
                                               int n) {
    int i = blockIdx.x * 256 + threadIdx.x;
    if (i < n) out[i] = f2bf(x[i]);
}

// W[k][n] = (wpn[k][n] - wpn[k+K][n]) * scale^(1/3), bf16, row-major K x N.
__global__ void __launch_bounds__(256) k_prep_w(const float* __restrict__ wpn,
                                                const float* __restrict__ scale,
                                                unsigned short* __restrict__ out,
                                                int total) {
    int i = blockIdx.x * 256 + threadIdx.x;
    if (i >= total) return;
    float ls = powf(scale[0], 1.0f / 3.0f);
    out[i] = f2bf((wpn[i] - wpn[i + total]) * ls);
}

// C = A(bf16, MxK) @ W(bf16, KxN) + bias, then either:
//   do_act: monotone pair activation -> Hout (bf16, M x N/2)
//   else  : Fout (fp32, M x N)
__global__ void __launch_bounds__(256) k_gemm(
    const unsigned short* __restrict__ A,
    const unsigned short* __restrict__ Wt,
    const float* __restrict__ bias,
    const float* __restrict__ act,
    unsigned short* __restrict__ Hout,
    float* __restrict__ Fout,
    int M, int N, int K, int do_act)
{
    constexpr int BM = 256, BN = 128, BK = 32;
    constexpr int LDA = 40;    // 32 + 8 pad  (80 B rows: odd multiple of 16 B)
    constexpr int LDB = 136;   // 128 + 8 pad (272 B rows: odd multiple of 16 B)
#if USE_ASYNC_LDS
    __shared__ __align__(16) unsigned short As[2][BM * LDA];   // 2 x 20 KB
    __shared__ __align__(16) unsigned short Bs[2][BK * LDB];   // 2 x 8.5 KB
#else
    __shared__ __align__(16) unsigned short As[1][BM * LDA];
    __shared__ __align__(16) unsigned short Bs[1][BK * LDB];
#endif

    const int tid  = threadIdx.x;
    const int lane = tid & 31;
    const int wave = tid >> 5;
    const int wm   = wave & 3;      // 4 waves along M (64 rows each)
    const int wn   = wave >> 2;     // 2 waves along N (64 cols each)
    const int mblock = blockIdx.y * BM;
    const int nblock = blockIdx.x * BN;
    const int KT = K / BK;

    v8f acc[4][4];
#pragma unroll
    for (int m = 0; m < 4; m++)
#pragma unroll
        for (int n = 0; n < 4; n++)
#pragma unroll
            for (int r = 0; r < 8; r++) acc[m][n][r] = 0.0f;

    // ---- compute step: fragments from LDS buffer `buf`, 16 WMMAs ----
    auto mma_step = [&](int buf) {
        v16bf afrag[4], bfrag[4];
        const int ks = (lane >> 4) * 8;   // A: lane 0-15 -> K 0..7/16..23, lane 16-31 -> K 8..15/24..31
#pragma unroll
        for (int m = 0; m < 4; m++) {
            const int arow = wm * 64 + m * 16 + (lane & 15);
            reinterpret_cast<uint4*>(&afrag[m])[0] = *reinterpret_cast<const uint4*>(&As[buf][arow * LDA + ks]);
            reinterpret_cast<uint4*>(&afrag[m])[1] = *reinterpret_cast<const uint4*>(&As[buf][arow * LDA + ks + 16]);
        }
#pragma unroll
        for (int n = 0; n < 4; n++) {     // B: lane&15 = N col, K in same order as A
            const int bn = wn * 64 + n * 16 + (lane & 15);
#pragma unroll
            for (int s = 0; s < 16; s++) {
                const int kk = (s < 8) ? (ks + s) : (16 + ks + (s - 8));
                bfrag[n][s] = *reinterpret_cast<const __bf16*>(&Bs[buf][kk * LDB + bn]);
            }
        }
#pragma unroll
        for (int m = 0; m < 4; m++)
#pragma unroll
            for (int n = 0; n < 4; n++)
                acc[m][n] = __builtin_amdgcn_wmma_f32_16x16x32_bf16(
                    false, afrag[m], false, bfrag[n], (short)0, acc[m][n], false, false);
    };

#if USE_ASYNC_LDS
    // ---- CDNA5 async copy path: global_load_async_to_lds_b128, ASYNCcnt ----
    auto issue_tile = [&](int k0, int buf) {
#pragma unroll
        for (int i = 0; i < 4; i++) {                 // A tile: 256x32 bf16 = 16 KB
            int ch = tid + i * 256;
            int row = ch >> 2, kc = (ch & 3) * 8;
            __builtin_amdgcn_global_load_async_to_lds_b128(
                (g_v4i*)(A + (size_t)(mblock + row) * K + k0 + kc),
                (l_v4i*)(&As[buf][row * LDA + kc]), 0, 0);
        }
#pragma unroll
        for (int i = 0; i < 2; i++) {                 // B tile: 32x128 bf16 = 8 KB
            int ch = tid + i * 256;
            int row = ch >> 4, nc = (ch & 15) * 8;
            __builtin_amdgcn_global_load_async_to_lds_b128(
                (g_v4i*)(Wt + (size_t)(k0 + row) * N + nblock + nc),
                (l_v4i*)(&Bs[buf][row * LDB + nc]), 0, 0);
        }
    };

    issue_tile(0, 0);
    for (int kt = 0; kt < KT; ++kt) {
        asm volatile("s_wait_asynccnt 0x0" ::: "memory");   // my writes for tile kt landed
        __syncthreads();                                    // everyone's writes visible; prev reads done
        if (kt + 1 < KT) issue_tile((kt + 1) * BK, (kt + 1) & 1);
        if (kt + 2 < KT) {
            __builtin_prefetch(Wt + (size_t)((kt + 2) * BK + (tid >> 4)) * N + nblock + (tid & 15) * 8, 0, 1);
            __builtin_prefetch(A + (size_t)(mblock + (tid >> 1)) * K + (kt + 2) * BK + (tid & 1) * 16, 0, 1);
        }
        mma_step(kt & 1);
    }
#else
    // ---- fallback: register staging, single LDS buffer ----
    uint4 ra[4], rb[2];
    auto g_load = [&](int k0) {
#pragma unroll
        for (int i = 0; i < 4; i++) {
            int ch = tid + i * 256;
            int row = ch >> 2, kc = (ch & 3) * 8;
            ra[i] = *reinterpret_cast<const uint4*>(A + (size_t)(mblock + row) * K + k0 + kc);
        }
#pragma unroll
        for (int i = 0; i < 2; i++) {
            int ch = tid + i * 256;
            int row = ch >> 4, nc = (ch & 15) * 8;
            rb[i] = *reinterpret_cast<const uint4*>(Wt + (size_t)(k0 + row) * N + nblock + nc);
        }
    };
    auto s_store = [&]() {
#pragma unroll
        for (int i = 0; i < 4; i++) {
            int ch = tid + i * 256;
            int row = ch >> 2, kc = (ch & 3) * 8;
            *reinterpret_cast<uint4*>(&As[0][row * LDA + kc]) = ra[i];
        }
#pragma unroll
        for (int i = 0; i < 2; i++) {
            int ch = tid + i * 256;
            int row = ch >> 4, nc = (ch & 15) * 8;
            *reinterpret_cast<uint4*>(&Bs[0][row * LDB + nc]) = rb[i];
        }
    };

    g_load(0);
    s_store();
    __syncthreads();
    for (int kt = 0; kt < KT; ++kt) {
        const bool more = (kt + 1 < KT);
        if (more) g_load((kt + 1) * BK);
        if (kt + 2 < KT) {
            __builtin_prefetch(Wt + (size_t)((kt + 2) * BK + (tid >> 4)) * N + nblock + (tid & 15) * 8, 0, 1);
            __builtin_prefetch(A + (size_t)(mblock + (tid >> 1)) * K + (kt + 2) * BK + (tid & 1) * 16, 0, 1);
        }
        mma_step(0);
        __syncthreads();
        if (more) s_store();
        __syncthreads();
    }
#endif

    // Epilogue. C layout: VGPR r -> M = r + 8*(lane>>4), N = lane&15.
#pragma unroll
    for (int n = 0; n < 4; n++) {
        const int col = nblock + wn * 64 + n * 16 + (lane & 15);
        const float bv = bias[col];
        const int g = col >> 1;
        float p1 = 0.f, p2 = 0.f, p3 = 0.f;
        if (do_act) { p1 = act[g * 4 + 1]; p2 = act[g * 4 + 2]; p3 = act[g * 4 + 3]; }
#pragma unroll
        for (int m = 0; m < 4; m++) {
#pragma unroll
            for (int r = 0; r < 8; r++) {
                const int row = mblock + wm * 64 + m * 16 + 8 * (lane >> 4) + r;
                float v = acc[m][n][r] + bv;
                if (do_act) {
                    // pair (2g, 2g+1) sits in adjacent lanes
                    float o  = __shfl_xor(v, 1, 32);
                    float a0 = (lane & 1) ? o : v;
                    float a1 = (lane & 1) ? v : o;
                    float mn = fminf(a0, a1);
                    float d  = fmaxf(a0, a1) - mn;
                    float ov = p3 * mn + ((a0 <= a1) ? p2 : p1) * d;
                    if ((lane & 1) == 0)
                        Hout[(size_t)row * (N >> 1) + g] = f2bf(ov);
                } else {
                    Fout[(size_t)row * N + col] = v;
                }
            }
        }
    }
}

extern "C" void kernel_launch(void* const* d_in, const int* in_sizes, int n_in,
                              void* d_out, int out_size, void* d_ws, size_t ws_size,
                              hipStream_t stream) {
    (void)in_sizes; (void)n_in; (void)out_size; (void)ws_size;
    const float* X    = (const float*)d_in[0];
    const float* sc   = (const float*)d_in[1];
    const float* wpn0 = (const float*)d_in[2];
    const float* b0   = (const float*)d_in[3];
    const float* act0 = (const float*)d_in[4];
    const float* wpn1 = (const float*)d_in[5];
    const float* b1   = (const float*)d_in[6];
    const float* act1 = (const float*)d_in[7];
    const float* wpn2 = (const float*)d_in[8];
    const float* b2   = (const float*)d_in[9];
    float* out = (float*)d_out;

    const int B = 8192, W = 2048;

    char* ws = (char*)d_ws;
    size_t off = 0;
    auto carve = [&](size_t bytes) -> char* {
        char* p = ws + off;
        off = (off + bytes + 255) & ~(size_t)255;
        return p;
    };
    unsigned short* Abf = (unsigned short*)carve((size_t)B * W * 2);       // 32 MB
    unsigned short* Hbf = (unsigned short*)carve((size_t)B * W * 2);       // 32 MB
    unsigned short* W0  = (unsigned short*)carve((size_t)W * 2 * W * 2);   // 16 MB
    unsigned short* W1  = (unsigned short*)carve((size_t)W * 2 * W * 2);   // 16 MB
    unsigned short* W2  = (unsigned short*)carve((size_t)W * W * 2);       //  8 MB

    k_cvt_x <<<(B * W) / 256, 256, 0, stream>>>(X, Abf, B * W);
    k_prep_w<<<(W * 2 * W) / 256, 256, 0, stream>>>(wpn0, sc, W0, W * 2 * W);
    k_prep_w<<<(W * 2 * W) / 256, 256, 0, stream>>>(wpn1, sc, W1, W * 2 * W);
    k_prep_w<<<(W * W) / 256, 256, 0, stream>>>(wpn2, sc, W2, W * W);

    dim3 blk(256);
    dim3 g01(2 * W / 128, B / 256);   // 32 x 32
    // layer 0: h = act0(X @ W0 + b0)  -> Hbf
    k_gemm<<<g01, blk, 0, stream>>>(Abf, W0, b0, act0, Hbf, nullptr, B, 2 * W, W, 1);
    // layer 1: h = act1(h @ W1 + b1)  -> Abf (X no longer needed)
    k_gemm<<<g01, blk, 0, stream>>>(Hbf, W1, b1, act1, Abf, nullptr, B, 2 * W, W, 1);
    // layer 2: out = h @ W2 + b2 (fp32)
    dim3 g2(W / 128, B / 256);        // 16 x 32
    k_gemm<<<g2, blk, 0, stream>>>(Abf, W2, b2, nullptr, nullptr, out, B, W, W, 0);
}
